// NeuralMMU_18975165514080
// MI455X (gfx1250) — matrix operand order repaired and checked by reading, not compiled, not run on hardware
//
#include <hip/hip_runtime.h>
#include <cmath>

typedef _Float16 v16h __attribute__((ext_vector_type(16)));
typedef _Float16 v8h  __attribute__((ext_vector_type(8)));
typedef float    v8f  __attribute__((ext_vector_type(8)));

#define WAVES_PER_BLOCK 8
#define THREADS (WAVES_PER_BLOCK * 32)

// Fast tanh using the CDNA5 hardware TRANS32 instruction v_tanh_f32.
__device__ __forceinline__ float tanh_hw(float y) {
#if __has_builtin(__builtin_amdgcn_tanhf)
    return __builtin_amdgcn_tanhf(y);
#else
    float r;
    asm volatile("v_tanh_f32 %0, %1\n\tv_nop" : "=v"(r) : "v"(y));
    return r;
#endif
}

// Build one packed-dword of two f16 halves from bits k0 and k0+1 of w.
// Bit set -> half = 2.0h (0x4000): single-bit constant, so this folds to
// shl/shr + and per half (k0 is a compile-time constant under full unroll).
__device__ __forceinline__ unsigned bitpair_h2(unsigned w, int k0) {
    unsigned lo = (k0 <= 14) ? ((w << (14 - k0)) & 0x00004000u)
                             : ((w >> (k0 - 14)) & 0x00004000u);
    int k1 = k0 + 1;
    unsigned hi = (k1 <= 30) ? ((w << (30 - k1)) & 0x40000000u)
                             : ((w >> (k1 - 30)) & 0x40000000u);
    return lo | hi;
}

__global__ __launch_bounds__(THREADS) void neural_mmu_kernel(
    const long long* __restrict__ vaddr,
    const float* __restrict__ W1, const float* __restrict__ b1,
    const float* __restrict__ W2, const float* __restrict__ b2,
    float* __restrict__ out, int ntiles)
{
    // Packed WMMA B-operand layouts (lane = column N, 16 K halfs per lane in AMD table order)
    __shared__ alignas(32) _Float16 w1p[8192];                 // [ks(2)][nt(8)][lane(32)][16]
    __shared__ alignas(32) _Float16 w2p[8192];                 // [kb(4)][n2(4)][lane(32)][16]
    __shared__ alignas(32) _Float16 hbuf[WAVES_PER_BLOCK][16 * 128]; // per-wave 16x128 H tile

    const int tid = threadIdx.x;

    // ---- one-time weight repack into WMMA operand layout ----
    // W1 is pre-scaled by 0.5: A-operand bits are encoded as 2.0h, so the
    // f32-accumulated products (2*b)*(w/2) == b*w exactly.
    for (int idx = tid; idx < 8192; idx += THREADS) {
        int j  = idx & 15;
        int ln = (idx >> 4) & 31;
        int nt = (idx >> 9) & 7;
        int ks = (idx >> 12) & 1;
        int K  = ks * 32 + ((ln & 16) ? 8 : 0) + j + ((j >= 8) ? 8 : 0);
        int N  = nt * 16 + (ln & 15);
        w1p[idx] = (_Float16)(0.5f * W1[K * 128 + N]);
    }
    for (int idx = tid; idx < 8192; idx += THREADS) {
        int j  = idx & 15;
        int ln = (idx >> 4) & 31;
        int n2 = (idx >> 9) & 3;
        int kb = (idx >> 11) & 3;
        int K  = kb * 32 + ((ln & 16) ? 8 : 0) + j + ((j >= 8) ? 8 : 0);
        int N  = n2 * 16 + (ln & 15);
        w2p[idx] = (_Float16)W2[K * 64 + N];
    }
    __syncthreads();

    const int lane = tid & 31;
    const int warp = tid >> 5;
    const int alo  = (lane < 16) ? 0 : 8;   // A-operand K sub-block offset
    const int mrow = lane & 15;             // column N (C layout) / row M (A layout)
    const int mhi  = (lane < 16) ? 0 : 8;   // C-layout row offset for upper lane half

    float bias1[8];
#pragma unroll
    for (int nt = 0; nt < 8; ++nt) bias1[nt] = b1[nt * 16 + mrow];
    float bias2[2];
#pragma unroll
    for (int n2 = 0; n2 < 2; ++n2) bias2[n2] = b2[n2 * 16 + mrow];

    const v16h* w1v = (const v16h*)w1p;
    const v16h* w2v = (const v16h*)w2p;
    _Float16*   hrow = &hbuf[warp][0];

    const int wave_id     = blockIdx.x * WAVES_PER_BLOCK + warp;
    const int wave_stride = gridDim.x * WAVES_PER_BLOCK;

    for (int t = wave_id; t < ntiles; t += wave_stride) {
        // each lane-half loads the 16 addresses of this tile
        unsigned long long addr =
            (unsigned long long)vaddr[(long long)t * 16 + mrow];
        unsigned wlo = (unsigned)addr;          // bits 0..31  -> K-step 0
        unsigned whi = (unsigned)(addr >> 32);  // bits 32..63 -> K-step 1

        // ---- build layer-1 A operands (16x32 f16, doc layout) from address bits ----
        v16h a0, a1;
        unsigned* a0u = (unsigned*)&a0;
        unsigned* a1u = (unsigned*)&a1;
#pragma unroll
        for (int d = 0; d < 8; ++d) {                 // dword d holds halfs 2d, 2d+1
            int j0 = 2 * d;
            int k0 = alo + j0 + ((j0 >= 8) ? 8 : 0);  // K of low half; high half is k0+1
            a0u[d] = bitpair_h2(wlo, k0);
            a1u[d] = bitpair_h2(whi, k0);
        }

        // ---- layer 1: [16x64] x [64x128] via 8 N-tiles x 2 K-steps ----
#pragma unroll
        for (int nt = 0; nt < 8; ++nt) {
            v8f acc = {};
            v16h wk0 = w1v[(0 * 8 + nt) * 32 + lane];
            v16h wk1 = w1v[(1 * 8 + nt) * 32 + lane];
            acc = __builtin_amdgcn_wmma_f32_16x16x32_f16(false, a0, false, wk0,
                                                         (short)0, acc, false, false);
            acc = __builtin_amdgcn_wmma_f32_16x16x32_f16(false, a1, false, wk1,
                                                         (short)0, acc, false, false);

            // bias + tanh-GELU, software-pipelined in 3 phases so the 8
            // independent v_tanh_f32 ops fill each other's TRANS32 latency
            // slots (no v_nop padding needed).
            float hx[8], th[8];
#pragma unroll
            for (int g = 0; g < 8; ++g) {
                float x  = acc[g] + bias1[nt];
                float x2 = x * x;
                th[g] = x * fmaf(0.03567740813636141f, x2, 0.7978845608028654f);
                hx[g] = 0.5f * x;
            }
#pragma unroll
            for (int g = 0; g < 8; ++g)
                th[g] = tanh_hw(th[g]);
#pragma unroll
            for (int g = 0; g < 8; ++g) {
                float ge = fmaf(hx[g], th[g], hx[g]);
                hrow[(g + mhi) * 128 + nt * 16 + mrow] = (_Float16)ge;
            }
        }
        asm volatile("s_wait_dscnt 0" ::: "memory");  // H tile complete (wave-private)

        // ---- relayout H -> layer-2 A operands (two 16B ds loads per K-step) ----
        v16h a2[4];
#pragma unroll
        for (int kb = 0; kb < 4; ++kb) {
            v8h lo = *(const v8h*)&hrow[mrow * 128 + kb * 32 + alo];
            v8h hi = *(const v8h*)&hrow[mrow * 128 + kb * 32 + alo + 16];
            a2[kb] = __builtin_shufflevector(lo, hi,
                     0,1,2,3,4,5,6,7,8,9,10,11,12,13,14,15);
        }
        asm volatile("s_wait_dscnt 0" ::: "memory");  // A2 in regs; H LDS reusable

        // ---- layer 2: only N=0..31 needed (logits 0..25 consumed) ----
        unsigned short* bbuf = (unsigned short*)hrow;  // reuse H tile for bit staging
#pragma unroll
        for (int n2 = 0; n2 < 2; ++n2) {
            v8f acc = {};
#pragma unroll
            for (int kb = 0; kb < 4; ++kb) {
                v16h w = w2v[(kb * 4 + n2) * 32 + lane];
                acc = __builtin_amdgcn_wmma_f32_16x16x32_f16(false, a2[kb], false, w,
                                                             (short)0, acc, false, false);
            }
#pragma unroll
            for (int g = 0; g < 8; ++g) {
                float logit = acc[g] + bias2[n2];
                bbuf[(g + mhi) * 32 + n2 * 16 + mrow] = (logit > 0.5f) ? 1u : 0u;
            }
        }
        asm volatile("s_wait_dscnt 0" ::: "memory");  // bits visible to packing lanes

        // ---- pack low 26 bits per row, store as float ----
        if (lane < 16) {
            unsigned p = 0;
#pragma unroll
            for (int n = 0; n < 26; ++n)
                p |= ((unsigned)bbuf[mrow * 32 + n]) << n;
            out[(long long)t * 16 + mrow] = (float)p;
        }
        asm volatile("s_wait_dscnt 0" ::: "memory");  // pack reads done before LDS reuse
    }
}

extern "C" void kernel_launch(void* const* d_in, const int* in_sizes, int n_in,
                              void* d_out, int out_size, void* d_ws, size_t ws_size,
                              hipStream_t stream) {
    const long long* vaddr = (const long long*)d_in[0];
    const float*     W1    = (const float*)d_in[1];
    const float*     b1    = (const float*)d_in[2];
    const float*     W2    = (const float*)d_in[3];
    const float*     b2    = (const float*)d_in[4];
    float*           out   = (float*)d_out;

    int Bsz    = in_sizes[0];
    int ntiles = Bsz / 16;           // B = 1,048,576 -> 65,536 tiles
    int blocks = 1024;               // 8192 waves, 8 tiles per wave

    neural_mmu_kernel<<<blocks, THREADS, 0, stream>>>(vaddr, W1, b1, W2, b2, out, ntiles);
}
